// EntRelJointDecoder_21689584845626
// MI455X (gfx1250) — matrix operand order repaired and checked by reading, not compiled, not run on hardware
//
#include <hip/hip_runtime.h>
#include <hip/hip_bf16.h>
#include <math.h>
#include <stdint.h>

typedef __attribute__((ext_vector_type(16))) _Float16 v16h;
typedef __attribute__((ext_vector_type(8)))  _Float16 v8h;
typedef __attribute__((ext_vector_type(8)))  float    v8f;
typedef __attribute__((ext_vector_type(4)))  float    v4f;
typedef __attribute__((ext_vector_type(4)))  unsigned int u32x4;
typedef __attribute__((ext_vector_type(8)))  int      i32x8;
typedef __attribute__((ext_vector_type(4)))  int      i32x4;

#define WMMA_F16(a, b, c) \
  __builtin_amdgcn_wmma_f32_16x16x32_f16(false, (a), false, (b), (short)0, (c), false, false)

union FragA { v16h v; v8h h[2]; _Float16 e[16]; };
union AccU  { v8f  v; float f[8]; };

__device__ __forceinline__ float gelu_exact(float x) {
  return 0.5f * x * (1.0f + erff(x * 0.70710678118654752f));
}

// Problem sizes
constexpr int B_ = 4, S_ = 128, H_ = 768, M_ = 256, O_ = 8, V_ = 10;
constexpr int BS = B_ * S_; // 512

// Workspace layout (byte offsets; all 16B aligned)
constexpr size_t OFF_RH  = 0;                                   // f16 [512][768]  r
constexpr size_t OFF_WHT = OFF_RH  + (size_t)BS * H_ * 2;       // f16 [256][768]  Wh^T
constexpr size_t OFF_WTT = OFF_WHT + (size_t)M_ * H_ * 2;       // f16 [256][768]  Wt^T
constexpr size_t OFF_WVT = OFF_WTT + (size_t)M_ * H_ * 2;       // f16 [256][768]  Wv^T
constexpr size_t OFF_UH  = OFF_WVT + (size_t)M_ * H_ * 2;       // f16 [8][256][256] U
constexpr size_t OFF_PH  = OFF_UH  + (size_t)O_ * M_ * M_ * 2;  // f32 [512][256]  Ph + b_pair
constexpr size_t OFF_PT  = OFF_PH  + (size_t)BS * M_ * 4;       // f32 [512][256]  Pt
constexpr size_t OFF_VAL = OFF_PT  + (size_t)BS * M_ * 4;       // f16 [512][256]  gelu(value)
constexpr size_t OFF_W2T = OFF_VAL + (size_t)BS * M_ * 2;       // f16 [4][8][128][256] W2^T
constexpr size_t OFF_ACC = OFF_W2T + (size_t)B_ * O_ * S_ * M_ * 2; // f32 [4] loss accumulators

// ---- WMMA fragment loaders --------------------------------------------------
__device__ __forceinline__ v16h load_fragA(const _Float16* base, int ld, int lane, int k0) {
  int n  = lane & 15;
  int ko = k0 + ((lane >> 4) << 3);
  const _Float16* p = base + (size_t)n * ld + ko;
  FragA f;
  f.h[0] = *(const v8h*)(p);
  f.h[1] = *(const v8h*)(p + 16);
  return f.v;
}
__device__ __forceinline__ v16h load_fragB(const _Float16* baseT, int ld, int lane, int k0) {
  int n  = lane & 15;
  int ko = k0 + ((lane >> 4) << 3);
  const _Float16* p = baseT + (size_t)n * ld + ko;
  FragA f;
  f.h[0] = *(const v8h*)(p);
  f.h[1] = *(const v8h*)(p + 16);
  return f.v;
}

// ---- K0: dtype conversion + weight transposes + accumulator clear -----------
__global__ void k_convert(const float* __restrict__ r, const float* __restrict__ Wp,
                          const float* __restrict__ Wv, const float* __restrict__ Uf,
                          _Float16* rh, _Float16* whT, _Float16* wtT, _Float16* wvT,
                          _Float16* uh, float* accs) {
  int gid = blockIdx.x * blockDim.x + threadIdx.x;
  if (gid < 4) accs[gid] = 0.f;
  int i = gid;
  if (i < BS * H_) { rh[i] = (_Float16)r[i]; return; }
  i -= BS * H_;
  if (i < M_ * H_) { int m = i / H_, h = i % H_; whT[i] = (_Float16)Wp[h * M_ + m]; return; }
  i -= M_ * H_;
  if (i < M_ * H_) { int m = i / H_, h = i % H_; wtT[i] = (_Float16)Wp[(H_ + h) * M_ + m]; return; }
  i -= M_ * H_;
  if (i < M_ * H_) { int m = i / H_, h = i % H_; wvT[i] = (_Float16)Wv[h * M_ + m]; return; }
  i -= M_ * H_;
  if (i < O_ * M_ * M_) { uh[i] = (_Float16)Uf[i]; }
}

// ---- K1: Ph = r@Wh + b_pair, Pt = r@Wt, val = gelu(r@Wv + b_value) ----------
__global__ void k_proj(const _Float16* __restrict__ rh, const _Float16* __restrict__ whT,
                       const _Float16* __restrict__ wtT, const _Float16* __restrict__ wvT,
                       const float* __restrict__ b_pair, const float* __restrict__ b_value,
                       float* ph, float* pt, _Float16* valh) {
  int lane = threadIdx.x;
  int mt = blockIdx.x;   // 0..15 (m tile)
  int rt = blockIdx.y;   // 0..31 (row tile over B*S)
  int row0 = rt * 16;
  const _Float16* abase = rh + (size_t)row0 * H_;
  const _Float16* bh = whT + (size_t)(mt * 16) * H_;
  const _Float16* bt = wtT + (size_t)(mt * 16) * H_;
  const _Float16* bv = wvT + (size_t)(mt * 16) * H_;

  v8f z = {0.f, 0.f, 0.f, 0.f, 0.f, 0.f, 0.f, 0.f};
  AccU ch, ct, cv; ch.v = z; ct.v = z; cv.v = z;

  for (int kk = 0; kk < H_ / 32; ++kk) {
    int k0 = kk * 32;
    v16h a  = load_fragA(abase, H_, lane, k0);
    v16h fh = load_fragB(bh, H_, lane, k0);
    v16h ft = load_fragB(bt, H_, lane, k0);
    v16h fv = load_fragB(bv, H_, lane, k0);
    ch.v = WMMA_F16(a, fh, ch.v);
    ct.v = WMMA_F16(a, ft, ct.v);
    cv.v = WMMA_F16(a, fv, cv.v);
  }
  int mcol = mt * 16 + (lane & 15);
  int half = lane >> 4;
  float bpv = b_pair[mcol], bvv = b_value[mcol];
#pragma unroll
  for (int v = 0; v < 8; ++v) {
    int grow = row0 + v + half * 8;
    ph[(size_t)grow * M_ + mcol] = ch.f[v] + bpv;      // fold b_pair into Ph
    pt[(size_t)grow * M_ + mcol] = ct.f[v];
    valh[(size_t)grow * M_ + mcol] = (_Float16)gelu_exact(cv.f[v] + bvv);
  }
}

// ---- K2: W2^T[b,o,z,i] = sum_j value[b,z,j] * U[o,i,j] ----------------------
__global__ void k_w2(const _Float16* __restrict__ valh, const _Float16* __restrict__ uh,
                     _Float16* w2t) {
  int lane = threadIdx.x;
  int it = blockIdx.x;   // 0..15 (i tile)
  int zt = blockIdx.y;   // 0..7  (z tile)
  int bo = blockIdx.z;   // 0..31
  int b = bo >> 3, o = bo & 7;
  const _Float16* abase = valh + ((size_t)b * S_ + zt * 16) * M_;   // rows = z, K = j
  const _Float16* bbase = uh + ((size_t)o * M_ + it * 16) * M_;     // B^T rows = i, K = j

  v8f z = {0.f, 0.f, 0.f, 0.f, 0.f, 0.f, 0.f, 0.f};
  AccU c; c.v = z;
  for (int kk = 0; kk < M_ / 32; ++kk) {
    v16h a  = load_fragA(abase, M_, lane, kk * 32);
    v16h bb = load_fragB(bbase, M_, lane, kk * 32);
    c.v = WMMA_F16(a, bb, c.v);
  }
  int icol = it * 16 + (lane & 15);
  int half = lane >> 4;
  _Float16* wout = w2t + (size_t)(b * O_ + o) * S_ * M_;
#pragma unroll
  for (int v = 0; v < 8; ++v) {
    int zz = zt * 16 + v + half * 8;
    wout[(size_t)zz * M_ + icol] = (_Float16)c.f[v];
  }
}

// ---- K3: joint_score + element CE loss (wave per (b,x,y)) -------------------
__global__ void k_joint(const float* __restrict__ ph, const float* __restrict__ pt,
                        const float* __restrict__ Wf, const float* __restrict__ bf,
                        const int* __restrict__ jlab, const unsigned char* __restrict__ jmask,
                        float* jout, float* accs) {
  int t = blockIdx.x * blockDim.x + threadIdx.x;
  int wid = t >> 5;          // 0..65535 == (b,x,y) flat
  int lane = t & 31;
  int b = wid >> 14;
  int x = (wid >> 7) & 127;
  int y = wid & 127;
  const float* phr = ph + (size_t)(b * S_ + x) * M_;
  const float* ptr_ = pt + (size_t)(b * S_ + y) * M_;
  int m0 = lane * 8;
  float pr[8];
#pragma unroll
  for (int i = 0; i < 8; ++i) pr[i] = gelu_exact(phr[m0 + i] + ptr_[m0 + i]);
  float lg[10];
#pragma unroll
  for (int v = 0; v < 10; ++v) lg[v] = 0.f;
#pragma unroll
  for (int i = 0; i < 8; ++i) {
    const float* w = Wf + (size_t)(m0 + i) * V_;
#pragma unroll
    for (int v = 0; v < 10; ++v) lg[v] += pr[i] * w[v];
  }
#pragma unroll
  for (int v = 0; v < 10; ++v) {
#pragma unroll
    for (int off = 16; off > 0; off >>= 1) lg[v] += __shfl_xor(lg[v], off, 32);
    lg[v] += bf[v];
  }
  if (lane == 0) {
    float* jo = jout + (size_t)wid * V_;
    float mx = lg[0];
#pragma unroll
    for (int v = 1; v < 10; ++v) mx = fmaxf(mx, lg[v]);
    float se = 0.f;
#pragma unroll
    for (int v = 0; v < 10; ++v) { jo[v] = lg[v]; se += __expf(lg[v] - mx); }
    int lbl = jlab[wid];
    float sel = 0.f;
#pragma unroll
    for (int v = 0; v < 10; ++v) sel = (v == lbl) ? lg[v] : sel;
    float nll = mx + __logf(se) - sel;
    float m = jmask[wid] ? 1.f : 0.f;
    atomicAdd(&accs[0], nll * m);
    atomicAdd(&accs[1], m);
  }
}

// ---- K4: q_score GEMM, TDM-staged B in LDS, fused pair gelu, softmax + CE ---
// Block = 4 waves = 64 consecutive xy rows (single x) for one (b, z-tile).
// Wave 0 DMAs the 8 o-slices of W2^T (8 x 8KB contiguous) into LDS via the
// Tensor Data Mover while every wave builds its A fragments in registers.
__global__ void __launch_bounds__(128) k_qscore(
    const float* __restrict__ ph, const float* __restrict__ pt,
    const _Float16* __restrict__ w2t,
    const int* __restrict__ qlab, const unsigned char* __restrict__ qmask,
    float* qout, float* accs) {
  __shared__ _Float16 sh_b[O_ * 16 * M_];   // 64 KB: B tiles for all 8 o

  int tid = threadIdx.x;
  int lane = tid & 31;
  int w = tid >> 5;          // wave 0..3
  int zt = blockIdx.x;       // 0..7
  int rtg = blockIdx.y;      // 0..255 : 64-row group of the 16384 xy rows per b
  int b = blockIdx.z;        // 0..3
  int x = rtg >> 1;          // 64 | 128, so one x per group
  int y0b = (rtg & 1) * 64;
  int n = lane & 15, half = lane >> 4;

  // --- Wave 0: issue one TDM 1-D tile copy per o (contiguous 8 KB each) ---
  if (w == 0) {
#pragma unroll
    for (int o = 0; o < 8; ++o) {
      const _Float16* gsrc = w2t + ((size_t)(b * O_ + o) * S_ + zt * 16) * M_;
      unsigned long long ga = (unsigned long long)(size_t)gsrc;
      unsigned int laddr = (unsigned int)(size_t)&sh_b[o * 16 * M_]; // LDS byte offset (low 32 bits of flat)
      u32x4 g0 = { 1u,                                   // count = 1 valid descriptor
                   laddr,                                // lds_addr
                   (unsigned int)(ga & 0xffffffffull),   // global_addr[31:0]
                   (unsigned int)((ga >> 32) & 0x01ffffffull) | (2u << 30) }; // addr[56:32] | type=2
      i32x8 g1 = { (int)(1u << 16),        // data_size = 1 (2-byte elems); wg_mask = 0
                   (int)(4096u << 16),     // tensor_dim0 = 4096 (bits 79:48, low part)
                   (int)(1 << 16),         // tensor_dim0 hi = 0 | tensor_dim1 = 1
                   (int)(4096u << 16),     // tensor_dim1 hi = 0 | tile_dim0 = 4096
                   0,                      // tile_dim1 = 0 (1-D tile), tile_dim2 = 0
                   4096,                   // tensor_dim0_stride = 4096
                   0, 0 };
      i32x4 g2 = { 0, 0, 0, 0 };
      i32x4 g3 = { 0, 0, 0, 0 };
      i32x8 g4 = { 0, 0, 0, 0, 0, 0, 0, 0 };   // extra group (clang-23 6-arg form), zero-filled
      __builtin_amdgcn_tensor_load_to_lds(g0, g1, g2, g3, g4, 0);
    }
  }

  // --- All waves: build 8 A fragments (pair = gelu(Ph+Pt)) while DMA runs ---
  const float* phr  = ph + (size_t)(b * S_ + x) * M_;
  const float* ptr_ = pt + (size_t)(b * S_ + (y0b + w * 16 + n)) * M_;
  __builtin_prefetch(phr, 0, 3);
  __builtin_prefetch(ptr_, 0, 3);
  FragA afr[8];
#pragma unroll
  for (int kk = 0; kk < 8; ++kk) {
    int ko = kk * 32 + half * 8;
#pragma unroll
    for (int i = 0; i < 8; ++i)
      afr[kk].e[i] = (_Float16)gelu_exact(phr[ko + i] + ptr_[ko + i]);
#pragma unroll
    for (int i = 0; i < 8; ++i)
      afr[kk].e[8 + i] = (_Float16)gelu_exact(phr[ko + 16 + i] + ptr_[ko + 16 + i]);
  }

  if (w == 0) __builtin_amdgcn_s_wait_tensorcnt(0);
  __syncthreads();

  // --- WMMA main loop: B fragments from LDS only ---
  v8f z8 = {0.f, 0.f, 0.f, 0.f, 0.f, 0.f, 0.f, 0.f};
  AccU acc[8];
#pragma unroll
  for (int o = 0; o < 8; ++o) acc[o].v = z8;

  const _Float16* pB = &sh_b[n * M_];
#pragma unroll
  for (int kk = 0; kk < 8; ++kk) {
    int koB = kk * 32 + half * 8;
#pragma unroll
    for (int o = 0; o < 8; ++o) {
      const _Float16* pb = pB + o * (16 * M_) + koB;
      FragA bb;
      bb.h[0] = *(const v8h*)(pb);
      bb.h[1] = *(const v8h*)(pb + 16);
      acc[o].v = WMMA_F16(afr[kk].v, bb.v, acc[o].v);
    }
  }

  // --- Epilogue: lane-local softmax over o, store raw scores, CE loss ---
  int zz = zt * 16 + n;
  float lsum = 0.f, lcnt = 0.f;
#pragma unroll
  for (int v = 0; v < 8; ++v) {
    int yy = y0b + w * 16 + v + half * 8;
    size_t qi = (((size_t)b * S_ + x) * S_ + yy) * S_ + zz;
    float s[8];
#pragma unroll
    for (int o = 0; o < 8; ++o) s[o] = acc[o].f[v];
    float* qo = qout + qi * O_;
    v4f s0 = {s[0], s[1], s[2], s[3]};
    v4f s1 = {s[4], s[5], s[6], s[7]};
    *(v4f*)(qo)     = s0;
    *(v4f*)(qo + 4) = s1;
    float mx = s[0];
#pragma unroll
    for (int o = 1; o < 8; ++o) mx = fmaxf(mx, s[o]);
    float ps = 0.f; float p[8];
#pragma unroll
    for (int o = 0; o < 8; ++o) { p[o] = __expf(s[o] - mx); ps += p[o]; }
    float inv = 1.f / ps;
#pragma unroll
    for (int o = 0; o < 8; ++o) p[o] *= inv;
    float pm = p[0];
#pragma unroll
    for (int o = 1; o < 8; ++o) pm = fmaxf(pm, p[o]);
    float se = 0.f;
#pragma unroll
    for (int o = 0; o < 8; ++o) se += __expf(p[o] - pm);
    int lbl = qlab[qi];
    float sel = 0.f;
#pragma unroll
    for (int o = 0; o < 8; ++o) sel = (o == lbl) ? p[o] : sel;
    float nll = pm + __logf(se) - sel;
    float mk = qmask[qi] ? 1.f : 0.f;
    lsum += nll * mk; lcnt += mk;
  }
#pragma unroll
  for (int off = 16; off > 0; off >>= 1) {
    lsum += __shfl_xor(lsum, off, 32);
    lcnt += __shfl_xor(lcnt, off, 32);
  }
  if (lane == 0) { atomicAdd(&accs[2], lsum); atomicAdd(&accs[3], lcnt); }
}

// ---- K5: combine losses -----------------------------------------------------
__global__ void k_final(const float* __restrict__ accs, float* out) {
  if (threadIdx.x == 0 && blockIdx.x == 0)
    out[0] = accs[0] / fmaxf(accs[1], 1.f) + accs[2] / fmaxf(accs[3], 1.f);
}

extern "C" void kernel_launch(void* const* d_in, const int* in_sizes, int n_in,
                              void* d_out, int out_size, void* d_ws, size_t ws_size,
                              hipStream_t stream) {
  (void)in_sizes; (void)n_in; (void)out_size; (void)ws_size;
  const float* r            = (const float*)d_in[0];
  const int* jlab           = (const int*)d_in[1];
  const unsigned char* jmsk = (const unsigned char*)d_in[2];
  const int* qlab           = (const int*)d_in[3];
  const unsigned char* qmsk = (const unsigned char*)d_in[4];
  const float* Wp           = (const float*)d_in[5];
  const float* bp           = (const float*)d_in[6];
  const float* Wf           = (const float*)d_in[7];
  const float* bf           = (const float*)d_in[8];
  const float* Wv           = (const float*)d_in[9];
  const float* bv           = (const float*)d_in[10];
  const float* Uf           = (const float*)d_in[11];

  char* ws = (char*)d_ws;
  _Float16* rh   = (_Float16*)(ws + OFF_RH);
  _Float16* whT  = (_Float16*)(ws + OFF_WHT);
  _Float16* wtT  = (_Float16*)(ws + OFF_WTT);
  _Float16* wvT  = (_Float16*)(ws + OFF_WVT);
  _Float16* uh   = (_Float16*)(ws + OFF_UH);
  float*    phw  = (float*)(ws + OFF_PH);
  float*    ptw  = (float*)(ws + OFF_PT);
  _Float16* valh = (_Float16*)(ws + OFF_VAL);
  _Float16* w2t  = (_Float16*)(ws + OFF_W2T);
  float*    accs = (float*)(ws + OFF_ACC);

  float* out  = (float*)d_out;
  float* jout = out + 1;                                  // joint_score
  float* qout = out + 1 + (size_t)B_ * S_ * S_ * V_;      // q_score

  int convN = BS * H_ + 3 * M_ * H_ + O_ * M_ * M_;       // 1,507,328
  k_convert<<<dim3((convN + 255) / 256), 256, 0, stream>>>(r, Wp, Wv, Uf, rh, whT, wtT, wvT, uh, accs);
  k_proj<<<dim3(16, 32), 32, 0, stream>>>(rh, whT, wtT, wvT, bp, bv, phw, ptw, valh);
  k_w2<<<dim3(16, 8, 32), 32, 0, stream>>>(valh, uh, w2t);
  k_joint<<<dim3((65536 * 32) / 256), 256, 0, stream>>>(phw, ptw, Wf, bf, jlab, jmsk, jout, accs);
  k_qscore<<<dim3(8, 256, 4), 128, 0, stream>>>(phw, ptw, w2t, qlab, qmsk, qout, accs);
  k_final<<<1, 1, 0, stream>>>(accs, out);
}
